// GRUBaseGloVe_31138512896870
// MI455X (gfx1250) — matrix operand order repaired and checked by reading, compiled, not run on hardware
//
#include <hip/hip_runtime.h>

// ---------------- sizes ----------------
#define BB   128
#define TT   256
#define EE   300
#define EP   320          // E padded to multiple of 32
#define HH   256
#define GG   768          // 3*H
#define H2   512          // 2*H
#define MM   (BB*TT)      // 32768 rows

typedef __attribute__((ext_vector_type(16))) __bf16 v16bf;
typedef __attribute__((ext_vector_type(8)))  float  v8f;

union Frag16 {            // one 16-bit WMMA operand fragment (8 VGPRs/lane)
    v16bf v;
    uint4 u[2];
    unsigned short s[16];
};

__device__ __forceinline__ unsigned short f2bf(float f) {
    unsigned int u = __float_as_uint(f);
    u += 0x7FFFu + ((u >> 16) & 1u);      // round-to-nearest-even
    return (unsigned short)(u >> 16);
}
__device__ __forceinline__ float sigf(float x)  { return 1.0f / (1.0f + __expf(-x)); }
__device__ __forceinline__ float tanhfast(float x) { return 2.0f / (1.0f + __expf(-2.0f * x)) - 1.0f; }

// ---------------- weight convert / pad (fp32 -> bf16, zero-padded cols) ----
__global__ __launch_bounds__(256) void cvt_pad_kernel(const float* __restrict__ src,
                                                      unsigned short* __restrict__ dst,
                                                      int rows, int sc, int dc) {
    size_t i = (size_t)blockIdx.x * 256 + threadIdx.x;
    if (i >= (size_t)rows * dc) return;
    int r = (int)(i / dc), c = (int)(i % dc);
    dst[i] = (c < sc) ? f2bf(src[(size_t)r * sc + c]) : (unsigned short)0;
}

// ---------------- embedding gather -> bf16 padded rows ----------------------
__global__ __launch_bounds__(EP) void embed_kernel(const int* __restrict__ tok,
                                                   const float* __restrict__ emb,
                                                   unsigned short* __restrict__ X0) {
    int m = blockIdx.x;           // 0..MM-1  (m = b*T + t)
    int k = threadIdx.x;          // 0..EP-1
    int tv = tok[m];
    unsigned short v = 0;
    if (k < EE) v = f2bf(emb[(size_t)tv * EE + k]);
    X0[(size_t)m * EP + k] = v;
}

// ---------------- input projection GEMM: xg = X @ W^T + b ------------------
// X: [MM, K] (bf16 pre-packed, or fp32 converted on the fly). W: [GG, K] bf16 row-major.
// Out: xg [MM, GG] fp32. Wave computes a 16x64 tile (1 M-tile x 4 N-tiles).
template <bool ABF16>
__global__ __launch_bounds__(256) void gemm_xg_kernel(const unsigned short* __restrict__ Abf,
                                                      const float* __restrict__ Af, int K,
                                                      const unsigned short* __restrict__ W,
                                                      const float* __restrict__ bias,
                                                      float* __restrict__ C) {
    const int wid = threadIdx.x >> 5, lane = threadIdx.x & 31;
    const int laneLow = lane & 15, laneHi = lane >> 4;
    const int gw = blockIdx.x * 8 + wid;              // 24576 waves total
    const int mTile = gw / 12, ng = gw % 12;
    const int m0 = mTile * 16, nbase = ng * 64;
    const int row = m0 + laneLow;
    const int nk = K >> 5;

    v8f acc[4] = {};
    for (int kc = 0; kc < nk; ++kc) {
        Frag16 a;
        if constexpr (ABF16) {
            const unsigned short* ar = Abf + (size_t)row * K + kc * 32 + laneHi * 8;
            a.u[0] = *(const uint4*)(ar);
            a.u[1] = *(const uint4*)(ar + 16);
        } else {
            const float* ar = Af + (size_t)row * K + kc * 32 + laneHi * 8;
            float4 f0 = *(const float4*)(ar);
            float4 f1 = *(const float4*)(ar + 4);
            float4 f2 = *(const float4*)(ar + 16);
            float4 f3 = *(const float4*)(ar + 20);
            a.s[0]  = f2bf(f0.x); a.s[1]  = f2bf(f0.y); a.s[2]  = f2bf(f0.z); a.s[3]  = f2bf(f0.w);
            a.s[4]  = f2bf(f1.x); a.s[5]  = f2bf(f1.y); a.s[6]  = f2bf(f1.z); a.s[7]  = f2bf(f1.w);
            a.s[8]  = f2bf(f2.x); a.s[9]  = f2bf(f2.y); a.s[10] = f2bf(f2.z); a.s[11] = f2bf(f2.w);
            a.s[12] = f2bf(f3.x); a.s[13] = f2bf(f3.y); a.s[14] = f2bf(f3.z); a.s[15] = f2bf(f3.w);
        }
#pragma unroll
        for (int j = 0; j < 4; ++j) {
            const unsigned short* wr = W + (size_t)(nbase + j * 16 + laneLow) * K + kc * 32 + laneHi * 8;
            Frag16 b;
            b.u[0] = *(const uint4*)(wr);
            b.u[1] = *(const uint4*)(wr + 16);
            acc[j] = __builtin_amdgcn_wmma_f32_16x16x32_bf16(false, a.v, false, b.v,
                                                             (short)0, acc[j], false, false);
        }
    }
#pragma unroll
    for (int j = 0; j < 4; ++j) {
        int col = nbase + j * 16 + laneLow;
        float bb = bias[col];
#pragma unroll
        for (int r = 0; r < 8; ++r)
            C[(size_t)(m0 + r + 8 * laneHi) * GG + col] = acc[j][r] + bb;
    }
}

// ---------------- GRU recurrent scan (persistent per batch-chunk) ----------
// grid (8, 2): x = batch chunk of 16 rows, y = direction. 256 threads = 8 waves.
// Whh fragments (6 tiles x 8 k-chunks x 8 VGPRs = 384 VGPRs/lane) are preloaded
// into registers ONCE; each of the 256 steps is pure LDS A-loads + WMMA.
#define HBF_LD 264          // padded bf16 row stride
#define HG_LD  772          // padded fp32 row stride
#define SCAN_SMEM ((16*HG_LD + 16*HH) * 4 + 16*HBF_LD * 2)   // 74240 bytes

__global__ __launch_bounds__(256) void gru_scan_kernel(
        const float* __restrict__ xg,            // [2][MM][GG]
        const unsigned short* __restrict__ whh,  // [2][GG][HH] bf16
        const float* __restrict__ bhh,           // [2][GG]
        float* __restrict__ y) {                 // [MM][H2]
    extern __shared__ char smem_raw[];
    float*          hg  = (float*)smem_raw;                 // [16][HG_LD]
    float*          hfp = hg + 16 * HG_LD;                  // [16][HH]
    unsigned short* hbf = (unsigned short*)(hfp + 16 * HH); // [16][HBF_LD]

    const int dir = blockIdx.y;
    const int b0  = blockIdx.x * 16;
    const float*          xg_d  = xg  + (size_t)dir * ((size_t)MM * GG);
    const unsigned short* whh_d = whh + (size_t)dir * (GG * HH);
    const float*          bhh_d = bhh + dir * GG;

    const int tid = threadIdx.x;
    const int wid = tid >> 5, lane = tid & 31;
    const int laneLow = lane & 15, laneHi = lane >> 4;

    // ---- preload Whh fragments into registers (resident for whole scan) ----
    Frag16 bw[6][8];
    float  bb[6];
#pragma unroll
    for (int j = 0; j < 6; ++j) {
        const int col = (wid * 6 + j) * 16 + laneLow;
        const unsigned short* wr = whh_d + (size_t)col * HH;
#pragma unroll
        for (int kc = 0; kc < 8; ++kc) {
            bw[j][kc].u[0] = *(const uint4*)(wr + kc * 32 + laneHi * 8);
            bw[j][kc].u[1] = *(const uint4*)(wr + kc * 32 + laneHi * 8 + 16);
        }
        bb[j] = bhh_d[col];
    }

    for (int i = tid; i < 16 * HBF_LD; i += 256) hbf[i] = 0;
    for (int i = tid; i < 16 * HH;     i += 256) hfp[i] = 0.0f;
    __syncthreads();

    const unsigned short* hrow = hbf + laneLow * HBF_LD;

    for (int tt = 0; tt < TT; ++tt) {
        const int t = dir ? (TT - 1 - tt) : tt;

        // ---- phase 1: hg = h @ Whh^T + bhh ----
        v8f c[6];
#pragma unroll
        for (int j = 0; j < 6; ++j) c[j] = (v8f){};
#pragma unroll
        for (int kc = 0; kc < 8; ++kc) {
            Frag16 a;
            a.u[0] = *(const uint4*)(hrow + kc * 32 + laneHi * 8);
            a.u[1] = *(const uint4*)(hrow + kc * 32 + 16 + laneHi * 8);
#pragma unroll
            for (int j = 0; j < 6; ++j)
                c[j] = __builtin_amdgcn_wmma_f32_16x16x32_bf16(false, a.v, false, bw[j][kc].v,
                                                               (short)0, c[j], false, false);
        }
#pragma unroll
        for (int j = 0; j < 6; ++j) {
            const int col = (wid * 6 + j) * 16 + laneLow;
#pragma unroll
            for (int r = 0; r < 8; ++r)
                hg[(r + 8 * laneHi) * HG_LD + col] = c[j][r] + bb[j];
        }
        __syncthreads();

        // ---- phase 2: gate math; thread owns hidden index k = tid ----
        {
            const int k = tid;
            for (int b = 0; b < 16; ++b) {
                const int brow = b0 + b;
                const size_t xbase = ((size_t)brow * TT + t) * GG;
                float xr = xg_d[xbase + k];
                float xz = xg_d[xbase + HH + k];
                float xn = xg_d[xbase + 2 * HH + k];
                float hr = hg[b * HG_LD + k];
                float hz = hg[b * HG_LD + HH + k];
                float hn = hg[b * HG_LD + 2 * HH + k];
                float r = sigf(xr + hr);
                float z = sigf(xz + hz);
                float n = tanhfast(xn + r * hn);
                float hp = hfp[b * HH + k];
                float hnew = (1.0f - z) * n + z * hp;
                hfp[b * HH + k] = hnew;
                hbf[b * HBF_LD + k] = f2bf(hnew);
                y[((size_t)brow * TT + t) * H2 + dir * HH + k] = hnew;
            }
        }
        __syncthreads();
    }
}

// ---------------- final linear: out[128,18] = y[:,T-1,:] @ lin_w^T + b ----
__global__ __launch_bounds__(32) void final_linear_kernel(const float* __restrict__ y,
                                                          const float* __restrict__ lin_w,
                                                          const float* __restrict__ lin_b,
                                                          float* __restrict__ out) {
    int b = blockIdx.x;
    int o = threadIdx.x;
    if (o >= 18) return;
    const float* yr = y + ((size_t)b * TT + (TT - 1)) * H2;
    const float* wr = lin_w + (size_t)o * H2;
    float s = lin_b[o];
    for (int k = 0; k < H2; ++k) s += yr[k] * wr[k];
    out[b * 18 + o] = s;
}

// ---------------- host ----------------
extern "C" void kernel_launch(void* const* d_in, const int* in_sizes, int n_in,
                              void* d_out, int out_size, void* d_ws, size_t ws_size,
                              hipStream_t stream) {
    (void)in_sizes; (void)n_in; (void)out_size; (void)ws_size;
    const int*   overview = (const int*)  d_in[0];
    const float* emb      = (const float*)d_in[1];
    const float* w_ih0    = (const float*)d_in[2];   // [2,768,300]
    const float* w_hh0    = (const float*)d_in[3];   // [2,768,256]
    const float* b_ih0    = (const float*)d_in[4];   // [2,768]
    const float* b_hh0    = (const float*)d_in[5];   // [2,768]
    const float* w_ih     = (const float*)d_in[6];   // [2,2,768,512]
    const float* w_hh     = (const float*)d_in[7];   // [2,2,768,256]
    const float* b_ih     = (const float*)d_in[8];   // [2,2,768]
    const float* b_hh     = (const float*)d_in[9];   // [2,2,768]
    const float* lin_w    = (const float*)d_in[10];  // [18,512]
    const float* lin_b    = (const float*)d_in[11];  // [18]
    float* out = (float*)d_out;

    char* ws = (char*)d_ws;
    size_t off = 0;
    auto take = [&](size_t bytes) -> char* {
        char* p = ws + off;
        off += (bytes + 255) & ~(size_t)255;
        return p;
    };
    unsigned short* wih0_bf = (unsigned short*)take((size_t)2 * GG * EP * 2);
    unsigned short* whh0_bf = (unsigned short*)take((size_t)2 * GG * HH * 2);
    unsigned short* wih_bf  = (unsigned short*)take((size_t)4 * GG * H2 * 2);
    unsigned short* whh_bf  = (unsigned short*)take((size_t)4 * GG * HH * 2);
    unsigned short* X0      = (unsigned short*)take((size_t)MM * EP * 2);
    float*          xg      = (float*)take((size_t)2 * MM * GG * 4);
    float*          yA      = (float*)take((size_t)MM * H2 * 4);
    float*          yB      = (float*)take((size_t)MM * H2 * 4);
    const size_t MG = (size_t)MM * GG;

    // weight conversion (bf16, padded)
    cvt_pad_kernel<<<( (2*GG*EP) + 255) / 256, 256, 0, stream>>>(w_ih0, wih0_bf, 2 * GG, EE, EP);
    cvt_pad_kernel<<<( (2*GG*HH) + 255) / 256, 256, 0, stream>>>(w_hh0, whh0_bf, 2 * GG, HH, HH);
    cvt_pad_kernel<<<( (4*GG*H2) + 255) / 256, 256, 0, stream>>>(w_ih,  wih_bf,  4 * GG, H2, H2);
    cvt_pad_kernel<<<( (4*GG*HH) + 255) / 256, 256, 0, stream>>>(w_hh,  whh_bf,  4 * GG, HH, HH);

    // embedding gather -> bf16
    embed_kernel<<<MM, EP, 0, stream>>>(overview, emb, X0);

    const int GEMM_BLOCKS = (MM / 16) * 12 / 8;   // 3072

    // ---- layer 0 ----
    for (int dir = 0; dir < 2; ++dir)
        gemm_xg_kernel<true><<<GEMM_BLOCKS, 256, 0, stream>>>(
            X0, nullptr, EP,
            wih0_bf + (size_t)dir * GG * EP, b_ih0 + dir * GG, xg + dir * MG);
    gru_scan_kernel<<<dim3(8, 2), 256, SCAN_SMEM, stream>>>(xg, whh0_bf, b_hh0, yA);

    // ---- layer 1 ----
    for (int dir = 0; dir < 2; ++dir)
        gemm_xg_kernel<false><<<GEMM_BLOCKS, 256, 0, stream>>>(
            nullptr, yA, H2,
            wih_bf + (size_t)(0 * 2 + dir) * GG * H2, b_ih + (0 * 2 + dir) * GG, xg + dir * MG);
    gru_scan_kernel<<<dim3(8, 2), 256, SCAN_SMEM, stream>>>(
        xg, whh_bf + (size_t)0 * 2 * GG * HH, b_hh + 0 * 2 * GG, yB);

    // ---- layer 2 ----
    for (int dir = 0; dir < 2; ++dir)
        gemm_xg_kernel<false><<<GEMM_BLOCKS, 256, 0, stream>>>(
            nullptr, yB, H2,
            wih_bf + (size_t)(1 * 2 + dir) * GG * H2, b_ih + (1 * 2 + dir) * GG, xg + dir * MG);
    gru_scan_kernel<<<dim3(8, 2), 256, SCAN_SMEM, stream>>>(
        xg, whh_bf + (size_t)1 * 2 * GG * HH, b_hh + 1 * 2 * GG, yA);

    // ---- final linear ----
    final_linear_kernel<<<BB, 32, 0, stream>>>(yA, lin_w, lin_b, out);
}
